// GNN_Only_22308060135550
// MI455X (gfx1250) — matrix operand order repaired and checked by reading, not compiled, not run on hardware
//
#include <hip/hip_runtime.h>
#include <hip/hip_fp16.h>

// GNN pipeline for MI455X (gfx1250, wave32):
//  k0: zero BN accumulators + pre-pack MLP weights into WMMA B-matrix lane layout
//  k1: per-sample fused GCN (one wave32 per sample), h2 -> ws as fp16,
//      BN sum/sumsq via shfl_xor butterflies -> LDS -> global f32 atomics
//  k2: fold BN into per-node scale sc[n] and constant T
//  k3: WMMA f16 MLP head (16 samples / wave): 53->64->32 via v_wmma_f32_16x16x32_f16,
//      32->1 + sigmoid via VALU + cross-lane reduction

#define B_TOTAL 131072
#define NN 17
#define HH 32

typedef __attribute__((ext_vector_type(16))) _Float16 v16h;
typedef __attribute__((ext_vector_type(8)))  float    v8f;

__device__ __forceinline__ float elu1(float x) {
    return x > 0.f ? x : (__expf(x) - 1.f);
}

// ---------------- Kernel 0: init stats + prepack MLP weights --------------------
__global__ __launch_bounds__(256) void k0_init(
    const float* __restrict__ Wr1, const float* __restrict__ br1,
    const float* __restrict__ Wr2, const float* __restrict__ br2,
    const float* __restrict__ Wr3, const float* __restrict__ br3,
    float* __restrict__ gsum, float* __restrict__ gss,
    _Float16* __restrict__ w1p, _Float16* __restrict__ w2p,
    float* __restrict__ w3, float* __restrict__ b1p,
    float* __restrict__ b2p, float* __restrict__ b3p)
{
    int t = threadIdx.x;
    if (t < 32) { gsum[t] = 0.f; gss[t] = 0.f; }
    // Wr1 [53,64] -> padded [64,64], 4 N-tiles x 2 K-steps, B-matrix lane layout:
    // linear idx = (((nt*2+ks)*32 + lane)*16 + i); K = ks*32 + i + 16*(lane>=16); N = nt*16 + lane%16
    for (int idx = t; idx < 4096; idx += 256) {
        int i  = idx & 15;
        int ln = (idx >> 4) & 31;
        int ks = (idx >> 9) & 1;
        int nt = idx >> 10;
        int K = ks * 32 + i + 16 * (ln >> 4);
        int N = nt * 16 + (ln & 15);
        w1p[idx] = (_Float16)((K < 53) ? Wr1[K * 64 + N] : 0.f);
    }
    // Wr2 [64,32], 2 N-tiles x 2 K-steps
    for (int idx = t; idx < 2048; idx += 256) {
        int i  = idx & 15;
        int ln = (idx >> 4) & 31;
        int ks = (idx >> 9) & 1;
        int nt = idx >> 10;
        int K = ks * 32 + i + 16 * (ln >> 4);
        int N = nt * 16 + (ln & 15);
        w2p[idx] = (_Float16)Wr2[K * 32 + N];
    }
    if (t < 32) w3[t]  = Wr3[t];
    if (t < 64) b1p[t] = br1[t];
    if (t < 32) b2p[t] = br2[t];
    if (t == 0) b3p[0] = br3[0];
}

// ---------------- Kernel 1: fused per-sample GCN (1 wave32 / sample) -----------
__global__ __launch_bounds__(256) void k1_gcn(
    const float* __restrict__ x_str, const float* __restrict__ adj,
    const float* __restrict__ W1, const float* __restrict__ b1,
    const float* __restrict__ W2, const float* __restrict__ b2,
    _Float16* __restrict__ h2f, float* __restrict__ gsum, float* __restrict__ gss)
{
    constexpr int WAVES = 8;
    __shared__ float s_sup [WAVES][NN * NN];     // adj, then normalized support in place
    __shared__ float s_dinv[WAVES][NN];
    __shared__ float s_xs  [WAVES][NN];
    __shared__ float s_agg [WAVES][NN];
    __shared__ float s_h1  [WAVES][NN * HH];
    __shared__ float s_a2  [WAVES][NN * HH];
    __shared__ float s_bsum[NN], s_bss[NN];

    const int lane = threadIdx.x & 31;
    const int w    = threadIdx.x >> 5;
    const int b    = blockIdx.x * WAVES + w;

    if (threadIdx.x < NN) { s_bsum[threadIdx.x] = 0.f; s_bss[threadIdx.x] = 0.f; }
    __syncthreads();

    const float* A = adj + (size_t)b * (NN * NN);
    for (int idx = lane; idx < NN * NN; idx += 32) s_sup[w][idx] = A[idx];
    if (lane < NN) s_xs[w][lane] = x_str[(size_t)b * NN + lane];
    __builtin_amdgcn_wave_barrier();

    // degree + D^{-1/2}
    if (lane < NN) {
        float d = 1.f;
        for (int m = 0; m < NN; ++m) d += s_sup[w][lane * NN + m];
        s_dinv[w][lane] = d > 0.f ? rsqrtf(d) : 0.f;
    }
    __builtin_amdgcn_wave_barrier();

    // support = D^{-1/2} (A + I) D^{-1/2}, in place
    for (int idx = lane; idx < NN * NN; idx += 32) {
        int n = idx / NN, m = idx - n * NN;
        float a = s_sup[w][idx] + (n == m ? 1.f : 0.f);
        s_sup[w][idx] = s_dinv[w][n] * s_dinv[w][m] * a;
    }
    __builtin_amdgcn_wave_barrier();

    // aggx = support @ x_str            [17]
    if (lane < NN) {
        float acc = 0.f;
        for (int m = 0; m < NN; ++m) acc += s_sup[w][lane * NN + m] * s_xs[w][m];
        s_agg[w][lane] = acc;
    }
    __builtin_amdgcn_wave_barrier();

    // h1[n][j] = elu(aggx[n]*W1[j] + b1[j])   (lane = j)
    const float w1j = W1[lane], b1j = b1[lane];
    for (int n = 0; n < NN; ++n)
        s_h1[w][n * HH + lane] = elu1(s_agg[w][n] * w1j + b1j);
    __builtin_amdgcn_wave_barrier();

    // agg2[n][k] = sum_m support[n][m] * h1[m][k]   (lane = k)
    for (int n = 0; n < NN; ++n) {
        float acc = 0.f;
        for (int m = 0; m < NN; ++m)
            acc += s_sup[w][n * NN + m] * s_h1[w][m * HH + lane];
        s_a2[w][n * HH + lane] = acc;
    }
    __builtin_amdgcn_wave_barrier();

    // W2 column j in registers
    float w2c[HH];
    #pragma unroll
    for (int k = 0; k < HH; ++k) w2c[k] = W2[k * HH + lane];
    const float b2j = b2[lane];

    _Float16* hout = h2f + (size_t)b * (NN * HH);
    for (int n = 0; n < NN; ++n) {
        float acc = b2j;
        #pragma unroll
        for (int k = 0; k < HH; ++k) acc += s_a2[w][n * HH + k] * w2c[k];
        float h = elu1(acc);
        hout[n * HH + lane] = (_Float16)h;
        // BatchNorm statistics (per node n, over batch x hidden)
        float s1 = h, s2 = h * h;
        #pragma unroll
        for (int off = 16; off; off >>= 1) {
            s1 += __shfl_xor(s1, off, 32);
            s2 += __shfl_xor(s2, off, 32);
        }
        if (lane == 0) { atomicAdd(&s_bsum[n], s1); atomicAdd(&s_bss[n], s2); }
    }
    __syncthreads();
    if (threadIdx.x < NN) {
        atomicAdd(&gsum[threadIdx.x], s_bsum[threadIdx.x]);
        atomicAdd(&gss [threadIdx.x], s_bss [threadIdx.x]);
    }
}

// ---------------- Kernel 2: fold BN into per-node scale + constant -------------
__global__ void k2_finalize(const float* __restrict__ gsum, const float* __restrict__ gss,
                            const float* __restrict__ gamma, const float* __restrict__ beta,
                            float* __restrict__ scT)
{
    __shared__ float tt[NN];
    int t = threadIdx.x;
    const float inv = 1.f / ((float)B_TOTAL * (float)HH);
    if (t < NN) {
        float mean = gsum[t] * inv;
        float var  = gss[t] * inv - mean * mean;
        float s    = gamma[t] * rsqrtf(var + 1e-5f);
        scT[t] = s * (1.f / (float)NN);          // folds mean-over-nodes
        tt[t]  = beta[t] - s * mean;
    }
    __syncthreads();
    if (t == 0) {
        float T = 0.f;
        for (int n = 0; n < NN; ++n) T += tt[n];
        scT[NN] = T * (1.f / (float)NN);
    }
}

// ---------------- Kernel 3: WMMA MLP head (16 samples per wave) ----------------
__global__ __launch_bounds__(256) void k3_mlp(
    const _Float16* __restrict__ h2f,
    const float* __restrict__ x_raw, const float* __restrict__ x_cov,
    const float* __restrict__ age, const float* __restrict__ scT,
    const _Float16* __restrict__ w1p, const _Float16* __restrict__ w2p,
    const float* __restrict__ w3, const float* __restrict__ b1p,
    const float* __restrict__ b2p, const float* __restrict__ b3p,
    float* __restrict__ out)
{
    constexpr int WAVES = 8;
    __shared__ float s_c[WAVES][16 * 64];        // combined features, then z1
    const int lane = threadIdx.x & 31;
    const int w    = threadIdx.x >> 5;
    const int b0   = (blockIdx.x * WAVES + w) * 16;
    const int am   = lane & 15;
    const int hi   = lane >> 4;

    float sc[NN];
    #pragma unroll
    for (int n = 0; n < NN; ++n) sc[n] = scT[n];
    const float T = scT[NN];

    // graph_emb -> s_c[m][0..31]
    for (int m = 0; m < 16; ++m) {
        const _Float16* hp = h2f + ((size_t)(b0 + m) * NN) * HH + lane;
        float acc = T;
        #pragma unroll
        for (int n = 0; n < NN; ++n) acc += sc[n] * (float)hp[n * HH];
        s_c[w][m * 64 + lane] = acc;
    }
    // raw features -> s_c[m][32..63] (x_raw 17, x_cov 3, age 1, zero pad)
    for (int m = 0; m < 16; ++m) {
        float v;
        if (lane < 17)       v = x_raw[(size_t)(b0 + m) * 17 + lane];
        else if (lane < 20)  v = x_cov[(size_t)(b0 + m) * 3 + (lane - 17)];
        else if (lane == 20) v = age[b0 + m];
        else                 v = 0.f;
        s_c[w][m * 64 + 32 + lane] = v;
    }
    __builtin_amdgcn_wave_barrier();

    // A operands (16x32 f16): lane = row M, halfword i -> K = i + 16*(lane>=16)
    v16h A0, A1;
    #pragma unroll
    for (int i = 0; i < 16; ++i) {
        int K = i + 16 * hi;
        A0[i] = (_Float16)s_c[w][am * 64 + K];
        A1[i] = (_Float16)s_c[w][am * 64 + 32 + K];
    }

    // Layer 1: [16,64] @ [64,64] -> ReLU -> z1 back into LDS
    const v16h* bp1 = (const v16h*)w1p;
    #pragma unroll
    for (int nt = 0; nt < 4; ++nt) {
        v16h B0 = bp1[(nt * 2 + 0) * 32 + lane];
        v16h B1 = bp1[(nt * 2 + 1) * 32 + lane];
        v8f c = {};
        c = __builtin_amdgcn_wmma_f32_16x16x32_f16(false, A0, false, B0, (short)0, c, false, false);
        c = __builtin_amdgcn_wmma_f32_16x16x32_f16(false, A1, false, B1, (short)0, c, false, false);
        int N = nt * 16 + am;
        float bias = b1p[N];
        #pragma unroll
        for (int r = 0; r < 8; ++r) {
            int M = r + 8 * hi;
            float z = c[r] + bias;
            s_c[w][M * 64 + N] = z > 0.f ? z : 0.f;
        }
    }
    __builtin_amdgcn_wave_barrier();

    // Layer 2: [16,64] @ [64,32]
    v16h A2, A3;
    #pragma unroll
    for (int i = 0; i < 16; ++i) {
        int K = i + 16 * hi;
        A2[i] = (_Float16)s_c[w][am * 64 + K];
        A3[i] = (_Float16)s_c[w][am * 64 + 32 + K];
    }
    const v16h* bp2 = (const v16h*)w2p;
    v8f c2[2];
    #pragma unroll
    for (int nt = 0; nt < 2; ++nt) {
        v16h B0 = bp2[(nt * 2 + 0) * 32 + lane];
        v16h B1 = bp2[(nt * 2 + 1) * 32 + lane];
        v8f c = {};
        c = __builtin_amdgcn_wmma_f32_16x16x32_f16(false, A2, false, B0, (short)0, c, false, false);
        c = __builtin_amdgcn_wmma_f32_16x16x32_f16(false, A3, false, B1, (short)0, c, false, false);
        c2[nt] = c;
    }

    // Layer 3 ([16,32] @ [32,1]) + sigmoid, in VALU with cross-lane reduce
    const float bias0 = b2p[am], bias1 = b2p[16 + am];
    const float w3a = w3[am], w3b = w3[16 + am];
    const float b3 = b3p[0];
    #pragma unroll
    for (int r = 0; r < 8; ++r) {
        float za = c2[0][r] + bias0; za = za > 0.f ? za : 0.f;
        float zb = c2[1][r] + bias1; zb = zb > 0.f ? zb : 0.f;
        float p = za * w3a + zb * w3b;
        p += __shfl_xor(p, 1, 32);
        p += __shfl_xor(p, 2, 32);
        p += __shfl_xor(p, 4, 32);
        p += __shfl_xor(p, 8, 32);
        if (am == 0) {
            int M = r + 8 * hi;
            out[b0 + M] = 1.f / (1.f + __expf(-(p + b3)));
        }
    }
}

// ---------------- launcher ------------------------------------------------------
extern "C" void kernel_launch(void* const* d_in, const int* in_sizes, int n_in,
                              void* d_out, int out_size, void* d_ws, size_t ws_size,
                              hipStream_t stream)
{
    (void)in_sizes; (void)n_in; (void)out_size; (void)ws_size;
    const float* x_str = (const float*)d_in[0];
    const float* x_raw = (const float*)d_in[1];
    const float* adj   = (const float*)d_in[2];
    const float* x_cov = (const float*)d_in[3];
    const float* age   = (const float*)d_in[4];
    const float* W1    = (const float*)d_in[5];
    const float* b1    = (const float*)d_in[6];
    const float* W2    = (const float*)d_in[7];
    const float* b2    = (const float*)d_in[8];
    const float* gamma = (const float*)d_in[9];
    const float* beta  = (const float*)d_in[10];
    const float* Wr1   = (const float*)d_in[11];
    const float* br1   = (const float*)d_in[12];
    const float* Wr2   = (const float*)d_in[13];
    const float* br2   = (const float*)d_in[14];
    const float* Wr3   = (const float*)d_in[15];
    const float* br3   = (const float*)d_in[16];
    float* out = (float*)d_out;

    char* ws = (char*)d_ws;
    const size_t OFF_STAT = (size_t)B_TOTAL * NN * HH * 2;   // 142,606,336 (256B aligned)
    _Float16* h2f  = (_Float16*)(ws);
    float*    gsum = (float*)   (ws + OFF_STAT);
    float*    gss  = (float*)   (ws + OFF_STAT + 128);
    float*    scT  = (float*)   (ws + OFF_STAT + 256);
    _Float16* w1p  = (_Float16*)(ws + OFF_STAT + 384);           // 8192 B
    _Float16* w2p  = (_Float16*)(ws + OFF_STAT + 384 + 8192);    // 4096 B
    float*    w3   = (float*)   (ws + OFF_STAT + 384 + 12288);   // 128 B
    float*    b1p  = (float*)   (ws + OFF_STAT + 384 + 12416);   // 256 B
    float*    b2p  = (float*)   (ws + OFF_STAT + 384 + 12672);   // 128 B
    float*    b3p  = (float*)   (ws + OFF_STAT + 384 + 12800);   // 4 B

    k0_init<<<1, 256, 0, stream>>>(Wr1, br1, Wr2, br2, Wr3, br3,
                                   gsum, gss, w1p, w2p, w3, b1p, b2p, b3p);
    k1_gcn<<<B_TOTAL / 8, 256, 0, stream>>>(x_str, adj, W1, b1, W2, b2, h2f, gsum, gss);
    k2_finalize<<<1, 32, 0, stream>>>(gsum, gss, gamma, beta, scT);
    k3_mlp<<<B_TOTAL / 128, 256, 0, stream>>>(h2f, x_raw, x_cov, age, scT,
                                              w1p, w2p, w3, b1p, b2p, b3p, out);
}
